// GPT_25838523252796
// MI455X (gfx1250) — matrix operand order repaired and checked by reading, not compiled, not run on hardware
//
#include <hip/hip_runtime.h>
#include <hip/hip_bf16.h>

// ---------------------------------------------------------------------------
// GPT-2 small forward pass on gfx1250 (MI455X).
// f16 WMMA (v_wmma_f32_16x16x32_f16) with fp32 accumulate everywhere;
// workgroup-tiled GEMMs stage A/B tiles into LDS with the Tensor Data Mover
// (tensor_load_to_lds + s_wait_tensorcnt); flash attention with WMMA.
// ---------------------------------------------------------------------------

typedef __attribute__((ext_vector_type(16))) _Float16     v16h;
typedef __attribute__((ext_vector_type(8)))  _Float16     v8h;
typedef __attribute__((ext_vector_type(8)))  float        v8f;
typedef __attribute__((ext_vector_type(4)))  unsigned int u32x4;
typedef __attribute__((ext_vector_type(4)))  int          i32x4;
typedef __attribute__((ext_vector_type(8)))  int          i32x8;

#define DEVINL __device__ __forceinline__

DEVINL v8f zero8() {
    v8f z;
#pragma unroll
    for (int i = 0; i < 8; ++i) z[i] = 0.0f;
    return z;
}

DEVINL v8f wmma_f16(v16h a, v16h b, v8f c) {
    // D = A(16x32 f16) x B(32x16 f16) + C(16x16 f32)
    return __builtin_amdgcn_wmma_f32_16x16x32_f16(
        /*neg_a=*/false, a, /*neg_b=*/false, b,
        /*c_mod=*/(short)0, c, /*reuse_a=*/false, /*reuse_b=*/false);
}

// A-fragment (16x32, MxK) from row-major f16 [rows x ld] at tile (m0,k0).
// Lane L: M = L&15 ; lanes 0-15 hold K = k0+0..15,
//         lanes 16-31 hold K = k0+16..31.
DEVINL v16h frag_a(const _Float16* __restrict__ base, int ld, int m0, int k0, int lane) {
    const int m  = m0 + (lane & 15);
    const int kb = k0 + ((lane >> 4) << 4);
    const _Float16* p = base + (size_t)m * ld + kb;
    v8h lo = *(const v8h*)(p);
    v8h hi = *(const v8h*)(p + 8);
    v16h r;
#pragma unroll
    for (int i = 0; i < 8; ++i) { r[i] = lo[i]; r[i + 8] = hi[i]; }
    return r;
}

// B-fragment (32x16, KxN) from [N x K] row-major f16 (transposed weight).
// Lane L: N = L&15 ; lanes 0-15 hold K = k0+0..15, lanes 16-31 K = k0+16..31.
DEVINL v16h frag_b(const _Float16* __restrict__ base, int ld, int n0, int k0, int lane) {
    const int n  = n0 + (lane & 15);
    const int kb = k0 + ((lane >> 4) << 4);
    const _Float16* p = base + (size_t)n * ld + kb;
    v8h lo = *(const v8h*)(p);
    v8h hi = *(const v8h*)(p + 8);
    v16h r;
#pragma unroll
    for (int i = 0; i < 8; ++i) { r[i] = lo[i]; r[i + 8] = hi[i]; }
    return r;
}

// B-fragment with inline fp32 -> f16 conversion from [N x K] fp32 (lm_head),
// zero-filling rows >= nvalid (N padded to a multiple of 64).
DEVINL v16h frag_b_f32(const float* __restrict__ base, int ld, int n0, int k0,
                       int lane, int nvalid) {
    const int n  = n0 + (lane & 15);
    const int kb = k0 + ((lane >> 4) << 4);
    v16h r;
    if (n < nvalid) {
        const float* p = base + (size_t)n * ld + kb;
#pragma unroll
        for (int i = 0; i < 16; ++i) r[i] = (_Float16)p[i];
    } else {
#pragma unroll
        for (int i = 0; i < 16; ++i) r[i] = (_Float16)0.0f;
    }
    return r;
}

// ---------------------------------------------------------------------------
// Tensor Data Mover: 2D f16 tile (tile_w x tile_h rows) from global -> LDS.
// D# packing per CDNA5 ISA:
//   group0: [1:0]=count=1, [63:32]=lds_addr, [120:64]=global_addr, [127:126]=2
//   group1: [17:16]=data_size(1 => 2B), [79:48]=tensor_dim0, [111:80]=tensor_dim1,
//           [127:112]=tile_dim0, [143:128]=tile_dim1, [207:160]=tensor_dim0_stride
// ---------------------------------------------------------------------------
#if __has_builtin(__builtin_amdgcn_tensor_load_to_lds)
#define USE_TDM 1
DEVINL void tdm_load_tile_f16(unsigned lds_addr, const _Float16* gptr,
                              int tensor_w, int tensor_rows, int tile_w,
                              int tile_rows, int stride_elems) {
    const unsigned long long ga = (unsigned long long)(uintptr_t)gptr;
    u32x4 g0;
    g0[0] = 1u;                                   // count=1 (valid user descriptor)
    g0[1] = lds_addr;                             // LDS byte address
    g0[2] = (unsigned)(ga & 0xffffffffu);         // global_addr[31:0]
    g0[3] = (unsigned)((ga >> 32) & 0x1ffffffu)   // global_addr[56:32]
          | (2u << 30);                           // type = 2 ("image")
    i32x8 g1;
    g1[0] = (1 << 16);                            // data_size = 1 -> 2 bytes
    g1[1] = (tensor_w & 0xffff) << 16;            // tensor_dim0[15:0]
    g1[2] = ((unsigned)tensor_w >> 16)            // tensor_dim0[31:16]
          | ((tensor_rows & 0xffff) << 16);       // tensor_dim1[15:0]
    g1[3] = ((unsigned)tensor_rows >> 16)         // tensor_dim1[31:16]
          | ((tile_w & 0xffff) << 16);            // tile_dim0
    g1[4] = (tile_rows & 0xffff);                 // tile_dim1 (tile_dim2 = 0)
    g1[5] = stride_elems;                         // tensor_dim0_stride[31:0]
    g1[6] = 0;                                    // stride hi / dim1_stride lo
    g1[7] = 0;
    i32x4 z4;
#pragma unroll
    for (int i = 0; i < 4; ++i) z4[i] = 0;
#if defined(__clang_major__) && __clang_major__ >= 23
    i32x8 z8;
#pragma unroll
    for (int i = 0; i < 8; ++i) z8[i] = 0;
    __builtin_amdgcn_tensor_load_to_lds(g0, g1, z4, z4, z8, 0);
#else
    __builtin_amdgcn_tensor_load_to_lds(g0, g1, z4, z4, 0);
#endif
}
#else
#define USE_TDM 0
#endif

// ---------------------------------------------------------------------------
// Elementwise / prep kernels
// ---------------------------------------------------------------------------

__global__ void embed_kernel(const int* __restrict__ idx, const float* __restrict__ wte,
                             const float* __restrict__ wpe, float* __restrict__ x,
                             int Tc, int Cc, int total) {
    int i = blockIdx.x * blockDim.x + threadIdx.x;
    if (i >= total) return;
    int c  = i % Cc;
    int mt = i / Cc;
    int t  = mt % Tc;
    int tok = idx[mt];
    x[i] = wte[(size_t)tok * Cc + c] + wpe[(size_t)t * Cc + c];
}

// src [K x N] fp32  ->  dst [N x K] f16
__global__ void cvt_transpose_kernel(const float* __restrict__ src,
                                     _Float16* __restrict__ dst, int K, int N) {
    int i = blockIdx.x * blockDim.x + threadIdx.x;
    if (i >= K * N) return;
    int n = i / K;
    int k = i - n * K;
    dst[i] = (_Float16)src[(size_t)k * N + n];
}

__global__ void __launch_bounds__(256) layernorm_kernel(
    const float* __restrict__ x, const float* __restrict__ w, const float* __restrict__ b,
    _Float16* __restrict__ out, int Cc) {
    __shared__ float sred[8];
    const int row = blockIdx.x;
    const int tid = threadIdx.x;
    const float* xr = x + (size_t)row * Cc;

    float s = 0.0f;
    for (int c = tid; c < Cc; c += 256) s += xr[c];
#pragma unroll
    for (int o = 16; o; o >>= 1) s += __shfl_xor(s, o, 32);
    if ((tid & 31) == 0) sred[tid >> 5] = s;
    __syncthreads();
    float tot = 0.0f;
#pragma unroll
    for (int i = 0; i < 8; ++i) tot += sred[i];
    const float mu = tot / (float)Cc;
    __syncthreads();

    float vs = 0.0f;
    for (int c = tid; c < Cc; c += 256) { float d = xr[c] - mu; vs += d * d; }
#pragma unroll
    for (int o = 16; o; o >>= 1) vs += __shfl_xor(vs, o, 32);
    if ((tid & 31) == 0) sred[tid >> 5] = vs;
    __syncthreads();
    float vtot = 0.0f;
#pragma unroll
    for (int i = 0; i < 8; ++i) vtot += sred[i];
    const float rs = rsqrtf(vtot / (float)Cc + 1e-5f);

    _Float16* orow = out + (size_t)row * Cc;
    for (int c = tid; c < Cc; c += 256)
        orow[c] = (_Float16)((xr[c] - mu) * rs * w[c] + b[c]);
}

// qkv [M x 3C] f16 -> q,k [B,H,T,HS] f16 ; v transposed to [B,H,HS,T] f16
__global__ void qkv_split_kernel(const _Float16* __restrict__ qkv,
                                 _Float16* __restrict__ q, _Float16* __restrict__ k,
                                 _Float16* __restrict__ vt,
                                 int Tc, int Cc, int Hc, int HSc, int total) {
    int i = blockIdx.x * blockDim.x + threadIdx.x;
    if (i >= total) return;
    int c  = i % Cc;
    int bt = i / Cc;
    int t  = bt % Tc;
    int b  = bt / Tc;
    int h  = c / HSc;
    int d  = c - h * HSc;
    size_t row = (size_t)bt * (3 * Cc);
    size_t qi  = (((size_t)(b * Hc + h)) * Tc + t) * HSc + d;
    q[qi] = qkv[row + c];
    k[qi] = qkv[row + Cc + c];
    vt[(((size_t)(b * Hc + h)) * HSc + d) * Tc + t] = qkv[row + 2 * Cc + c];
}

DEVINL float gelu_tanh(float v) {
    const float g = 0.7978845608028654f * (v + 0.044715f * v * v * v);
    return 0.5f * v * (1.0f + tanhf(g));
}

// ---------------------------------------------------------------------------
// Workgroup-tiled WMMA GEMM with TDM staging.
// 256 threads = 8 waves compute a 128(M) x 128(N) tile; per 32-wide K-step,
// wave 0 DMAs A(128x32) and B(128x32) f16 tiles into LDS (tensor_load_to_lds,
// s_wait_tensorcnt), then each wave computes a 32x64 sub-tile (8 WMMAs) from
// LDS fragments.  ~64 FLOP per byte of global traffic.
// mode 0: f16 store (+bias)   mode 1: f16 store, bias+GELU   mode 2: fp32 +=
// ---------------------------------------------------------------------------
__global__ void __launch_bounds__(256) gemm_tiled_kernel(
    const _Float16* __restrict__ A, const _Float16* __restrict__ Wt,
    const float* __restrict__ bias, _Float16* __restrict__ out16,
    float* __restrict__ out32, int M, int N, int K, int ostride, int mode) {
    __shared__ _Float16 tileA[128 * 32];
    __shared__ _Float16 tileB[128 * 32];
    const int tid  = threadIdx.x;
    const int lane = tid & 31;
    const int wv   = tid >> 5;        // 0..7
    const int tilesN = N >> 7;
    const int bm = blockIdx.x / tilesN;
    const int bn = blockIdx.x - bm * tilesN;
    const int m0 = bm << 7, n0 = bn << 7;
    const int wm = wv >> 1;           // 0..3 -> 32 M-rows each
    const int wn = wv & 1;            // 0..1 -> 64 N-cols each

    v8f acc[2][4];
#pragma unroll
    for (int i = 0; i < 2; ++i)
#pragma unroll
        for (int j = 0; j < 4; ++j) acc[i][j] = zero8();

#if USE_TDM
    const unsigned ldsA = (unsigned)(uintptr_t)(void*)tileA;
    const unsigned ldsB = (unsigned)(uintptr_t)(void*)tileB;
#endif

    for (int k0 = 0; k0 < K; k0 += 32) {
#if USE_TDM
        if (wv == 0) {                 // one TDM issue per tile (wave-level DMA)
            tdm_load_tile_f16(ldsA, A  + (size_t)m0 * K + k0, K, M - m0, 32, 128, K);
            tdm_load_tile_f16(ldsB, Wt + (size_t)n0 * K + k0, K, N - n0, 32, 128, K);
            __builtin_amdgcn_s_wait_tensorcnt(0);
        }
#else
        for (int i = tid; i < 128 * 32; i += 256) {
            int r = i >> 5, c = i & 31;
            tileA[i] = A[(size_t)(m0 + r) * K + k0 + c];
            tileB[i] = Wt[(size_t)(n0 + r) * K + k0 + c];
        }
#endif
        __syncthreads();
#pragma unroll
        for (int i = 0; i < 2; ++i) {
            const v16h a = frag_a(tileA, 32, (wm << 5) + (i << 4), 0, lane);
#pragma unroll
            for (int j = 0; j < 4; ++j) {
                const v16h b = frag_b(tileB, 32, (wn << 6) + (j << 4), 0, lane);
                acc[i][j] = wmma_f16(a, b, acc[i][j]);
            }
        }
        asm volatile("s_wait_dscnt 0" ::: "memory");
        __syncthreads();               // tiles consumed before next DMA overwrite
    }

    const int hi = lane >> 4, nn = lane & 15;
#pragma unroll
    for (int i = 0; i < 2; ++i) {
#pragma unroll
        for (int j = 0; j < 4; ++j) {
            const int n  = n0 + (wn << 6) + (j << 4) + nn;
            const float bv = bias ? bias[n] : 0.0f;
#pragma unroll
            for (int r = 0; r < 8; ++r) {
                const int m = m0 + (wm << 5) + (i << 4) + r + (hi << 3);
                float v = acc[i][j][r] + bv;
                if (mode == 1) v = gelu_tanh(v);
                if (mode == 2)      out32[(size_t)m * ostride + n] += v;
                else                out16[(size_t)m * ostride + n] = (_Float16)v;
            }
        }
    }
}

// ---------------------------------------------------------------------------
// Direct-from-global WMMA GEMM for the lm_head: B operand is fp32 lm_w [V,C]
// converted to f16 in the fragment loader (no 77MB staging copy).
// One wave computes a 16x64 tile; fp32 store with N guard (V=50257, pad 50304).
// ---------------------------------------------------------------------------
__global__ void __launch_bounds__(128) gemm_lm_kernel(
    const _Float16* __restrict__ A, const float* __restrict__ Wf32,
    float* __restrict__ out32, int M, int N, int K, int Nvalid, int ostride) {
    const int wave = blockIdx.x * (blockDim.x >> 5) + (threadIdx.x >> 5);
    const int lane = threadIdx.x & 31;
    const int tilesN = N >> 6;
    const int tm = wave / tilesN;
    const int tn = wave - tm * tilesN;
    if (tm * 16 >= M) return;  // wave-uniform
    const int m0 = tm * 16, n0 = tn * 64;

    v8f acc[4];
#pragma unroll
    for (int j = 0; j < 4; ++j) acc[j] = zero8();

    for (int k0 = 0; k0 < K; k0 += 32) {
        const v16h a = frag_a(A, K, m0, k0, lane);
#pragma unroll
        for (int j = 0; j < 4; ++j) {
            const int n = n0 + 16 * j + (lane & 15);
            if (n + 0 < Nvalid)        // prefetch next K-chunk of this row
                __builtin_prefetch(Wf32 + (size_t)n * K + k0 + 32, 0, 1);
            v16h b = frag_b_f32(Wf32, K, n0 + 16 * j, k0, lane, Nvalid);
            acc[j] = wmma_f16(a, b, acc[j]);
        }
    }

    const int hi = lane >> 4, nn = lane & 15;
#pragma unroll
    for (int j = 0; j < 4; ++j) {
        const int n = n0 + 16 * j + nn;
#pragma unroll
        for (int r = 0; r < 8; ++r) {
            const int m = m0 + r + (hi << 3);
            if (n < Nvalid) out32[(size_t)m * ostride + n] = acc[j][r];
        }
    }
}

// ---------------------------------------------------------------------------
// Flash attention: one wave per (b, h, 16-query tile). Online softmax,
// QK^T and P*V via WMMA; P re-layouted C-frag -> A-frag through LDS.
// ---------------------------------------------------------------------------
__global__ void __launch_bounds__(32) attn_kernel(
    const _Float16* __restrict__ q, const _Float16* __restrict__ k,
    const _Float16* __restrict__ vt, _Float16* __restrict__ y,
    int Hc, int Tc, int HSc, float scale) {
    __shared__ _Float16 ptile[16 * 32];
    const int lane  = threadIdx.x & 31;
    const int qtiles = Tc >> 4;
    const int qt = blockIdx.x % qtiles;
    const int bh = blockIdx.x / qtiles;
    const int h  = bh % Hc;
    const int b  = bh / Hc;
    const int q0 = qt << 4;

    const _Float16* qp = q  + (size_t)bh * Tc * HSc;
    const _Float16* kp = k  + (size_t)bh * Tc * HSc;
    const _Float16* vp = vt + (size_t)bh * HSc * Tc;  // [HS, T]

    const v16h qa0 = frag_a(qp, HSc, q0, 0,  lane);
    const v16h qa1 = frag_a(qp, HSc, q0, 32, lane);

    v8f o0 = zero8(), o1 = zero8(), o2 = zero8(), o3 = zero8();
    float mst[8], lst[8];
#pragma unroll
    for (int r = 0; r < 8; ++r) { mst[r] = -1e30f; lst[r] = 0.0f; }

    const int hi = lane >> 4, nn = lane & 15;
    const int nkb = (q0 + 16 + 31) >> 5;  // 32-key chunks
    for (int jb = 0; jb < nkb; ++jb) {
        const int j0 = jb << 5;
        v8f s0 = zero8(), s1 = zero8();
        s0 = wmma_f16(qa0, frag_b(kp, HSc, j0,      0,  lane), s0);
        s0 = wmma_f16(qa1, frag_b(kp, HSc, j0,      32, lane), s0);
        s1 = wmma_f16(qa0, frag_b(kp, HSc, j0 + 16, 0,  lane), s1);
        s1 = wmma_f16(qa1, frag_b(kp, HSc, j0 + 16, 32, lane), s1);

#pragma unroll
        for (int r = 0; r < 8; ++r) {
            const int qrow = q0 + r + (hi << 3);
            float a0 = (j0 + nn      > qrow) ? -1e30f : s0[r] * scale;
            float a1 = (j0 + 16 + nn > qrow) ? -1e30f : s1[r] * scale;
            float rm = fmaxf(a0, a1);
#pragma unroll
            for (int off = 1; off < 16; off <<= 1) rm = fmaxf(rm, __shfl_xor(rm, off, 32));
            const float mnew  = fmaxf(mst[r], rm);
            const float p0    = __expf(a0 - mnew);
            const float p1    = __expf(a1 - mnew);
            float rs = p0 + p1;
#pragma unroll
            for (int off = 1; off < 16; off <<= 1) rs += __shfl_xor(rs, off, 32);
            const float alpha = __expf(mst[r] - mnew);
            lst[r] = lst[r] * alpha + rs;
            mst[r] = mnew;
            o0[r] *= alpha; o1[r] *= alpha; o2[r] *= alpha; o3[r] *= alpha;
            const int row = r + (hi << 3);
            ptile[row * 32 + nn]      = (_Float16)p0;
            ptile[row * 32 + 16 + nn] = (_Float16)p1;
        }
        asm volatile("s_wait_dscnt 0" ::: "memory");
        const v16h pa = frag_a(ptile, 32, 0, 0, lane);
        o0 = wmma_f16(pa, frag_b(vp, Tc, 0,  j0, lane), o0);
        o1 = wmma_f16(pa, frag_b(vp, Tc, 16, j0, lane), o1);
        o2 = wmma_f16(pa, frag_b(vp, Tc, 32, j0, lane), o2);
        o3 = wmma_f16(pa, frag_b(vp, Tc, 48, j0, lane), o3);
        asm volatile("s_wait_dscnt 0" ::: "memory");
    }

    const int Ct = Hc * HSc;
#pragma unroll
    for (int r = 0; r < 8; ++r) {
        const float inv = 1.0f / lst[r];
        const int qrow = q0 + r + (hi << 3);
        _Float16* yr = y + ((size_t)(b * Tc + qrow)) * Ct + h * HSc;
        yr[nn]      = (_Float16)(o0[r] * inv);
        yr[16 + nn] = (_Float16)(o1[r] * inv);
        yr[32 + nn] = (_Float16)(o2[r] * inv);
        yr[48 + nn] = (_Float16)(o3[r] * inv);
    }
}

// ---------------------------------------------------------------------------
// Host-side orchestration
// ---------------------------------------------------------------------------
extern "C" void kernel_launch(void* const* d_in, const int* in_sizes, int n_in,
                              void* d_out, int out_size, void* d_ws, size_t ws_size,
                              hipStream_t stream) {
    (void)in_sizes; (void)n_in; (void)out_size; (void)ws_size;
    constexpr int Lc = 12, Hc = 12, Cc = 768, Vc = 50257, Bc = 4, Tc = 1024;
    constexpr int HSc = Cc / Hc;            // 64
    constexpr int C3 = 3 * Cc, C4 = 4 * Cc; // 2304, 3072
    constexpr int Mr = Bc * Tc;             // 4096 rows
    constexpr int Vpad = 50304;             // 786 * 64

    const int*   idx   = (const int*)d_in[0];
    const float* wte   = (const float*)d_in[1];
    const float* wpe   = (const float*)d_in[2];
    const float* ln1w  = (const float*)d_in[3];
    const float* ln1b  = (const float*)d_in[4];
    const float* attnw = (const float*)d_in[5];
    const float* attnb = (const float*)d_in[6];
    const float* projw = (const float*)d_in[7];
    const float* projb = (const float*)d_in[8];
    const float* ln2w  = (const float*)d_in[9];
    const float* ln2b  = (const float*)d_in[10];
    const float* fcw   = (const float*)d_in[11];
    const float* fcb   = (const float*)d_in[12];
    const float* fc2w  = (const float*)d_in[13];
    const float* fc2b  = (const float*)d_in[14];
    const float* lnfw  = (const float*)d_in[15];
    const float* lnfb  = (const float*)d_in[16];
    const float* lmw   = (const float*)d_in[17];
    float* out = (float*)d_out;

    // Workspace carve-up (256B aligned slices)
    char* base = (char*)d_ws;
    size_t off = 0;
    auto take = [&](size_t bytes) -> char* {
        char* p = base + off;
        off += (bytes + 255) & ~(size_t)255;
        return p;
    };
    _Float16* wq   = (_Float16*)take((size_t)C3 * Cc * 2);  // qkv weight [3C,C]
    _Float16* wp   = (_Float16*)take((size_t)Cc * Cc * 2);  // proj weight [C,C]
    _Float16* wf   = (_Float16*)take((size_t)C4 * Cc * 2);  // fc weight [4C,C]
    _Float16* wf2  = (_Float16*)take((size_t)Cc * C4 * 2);  // fc2 weight [C,4C]
    float*    x    = (float*)take((size_t)Mr * Cc * 4);
    _Float16* h16  = (_Float16*)take((size_t)Mr * Cc * 2);
    _Float16* qkv  = (_Float16*)take((size_t)Mr * C3 * 2);
    _Float16* q16  = (_Float16*)take((size_t)Mr * Cc * 2);
    _Float16* k16  = (_Float16*)take((size_t)Mr * Cc * 2);
    _Float16* vt16 = (_Float16*)take((size_t)Mr * Cc * 2);
    _Float16* y16  = (_Float16*)take((size_t)Mr * Cc * 2);
    _Float16* a16  = (_Float16*)take((size_t)Mr * C4 * 2);

    auto gemm = [&](const _Float16* A, const _Float16* Wt, const float* bias,
                    _Float16* o16, float* o32, int M, int N, int K,
                    int ostride, int mode) {
        int blocks = (M / 128) * (N / 128);
        gemm_tiled_kernel<<<blocks, 256, 0, stream>>>(A, Wt, bias, o16, o32,
                                                      M, N, K, ostride, mode);
    };

    // Token + position embedding
    embed_kernel<<<(Mr * Cc + 255) / 256, 256, 0, stream>>>(idx, wte, wpe, x, Tc, Cc, Mr * Cc);

    for (int l = 0; l < Lc; ++l) {
        // Convert this layer's weights to f16, transposed to [N,K]
        cvt_transpose_kernel<<<(Cc * C3 + 255) / 256, 256, 0, stream>>>(
            attnw + (size_t)l * Cc * C3, wq, Cc, C3);
        cvt_transpose_kernel<<<(Cc * Cc + 255) / 256, 256, 0, stream>>>(
            projw + (size_t)l * Cc * Cc, wp, Cc, Cc);
        cvt_transpose_kernel<<<(Cc * C4 + 255) / 256, 256, 0, stream>>>(
            fcw + (size_t)l * Cc * C4, wf, Cc, C4);
        cvt_transpose_kernel<<<(C4 * Cc + 255) / 256, 256, 0, stream>>>(
            fc2w + (size_t)l * C4 * Cc, wf2, C4, Cc);

        // Attention block
        layernorm_kernel<<<Mr, 256, 0, stream>>>(x, ln1w + l * Cc, ln1b + l * Cc, h16, Cc);
        gemm(h16, wq, attnb + (size_t)l * C3, qkv, nullptr, Mr, C3, Cc, C3, 0);
        qkv_split_kernel<<<(Mr * Cc + 255) / 256, 256, 0, stream>>>(
            qkv, q16, k16, vt16, Tc, Cc, Hc, HSc, Mr * Cc);
        attn_kernel<<<Bc * Hc * (Tc / 16), 32, 0, stream>>>(q16, k16, vt16, y16,
                                                            Hc, Tc, HSc, 0.125f);
        gemm(y16, wp, projb + (size_t)l * Cc, nullptr, x, Mr, Cc, Cc, Cc, 2);

        // MLP block
        layernorm_kernel<<<Mr, 256, 0, stream>>>(x, ln2w + l * Cc, ln2b + l * Cc, h16, Cc);
        gemm(h16, wf, fcb + (size_t)l * C4, a16, nullptr, Mr, C4, Cc, C4, 1);
        gemm(a16, wf2, fc2b + (size_t)l * Cc, nullptr, x, Mr, Cc, C4, Cc, 2);
    }

    // Final LN + lm_head (reads fp32 lm_w [V,C] directly, converts in-loader)
    layernorm_kernel<<<Mr, 256, 0, stream>>>(x, lnfw, lnfb, h16, Cc);
    {
        int waves  = (Mr / 16) * (Vpad / 64);
        int blocks = (waves + 3) / 4;
        gemm_lm_kernel<<<blocks, 128, 0, stream>>>(h16, lmw, out, Mr, Vpad, Cc, Vc, Vc);
    }
}
